// PTSegStacked_67714454389198
// MI455X (gfx1250) — matrix operand order repaired and checked by reading, not compiled, not run on hardware
//
#include <hip/hip_runtime.h>
#include <cstdint>

#define NPTS 65536
#define CH   64
#define NSZ  16
#define CSZ  8

typedef __attribute__((ext_vector_type(16))) __bf16  bf16x16;
typedef __attribute__((ext_vector_type(8)))  float   f32x8;

union Frag { bf16x16 v; unsigned u[8]; };

__device__ __forceinline__ unsigned bf16_rne(float f) {
  unsigned u = __float_as_uint(f);
  unsigned r = u + 0x7fffu + ((u >> 16) & 1u);
  return r >> 16;
}
__device__ __forceinline__ unsigned pack_bf16x2(float lo, float hi) {
  return bf16_rne(lo) | (bf16_rne(hi) << 16);
}

// ---------------------------------------------------------------------------
// Kernel A: xq/xk/xv = x @ {Wq,Wk,Wv} + bias via v_wmma_f32_16x16x32_bf16.
// Weights are bulk-copied into LDS with GLOBAL_LOAD_ASYNC_TO_LDS_B128 (no VGPR
// round trip, ASYNCcnt-tracked), then packed once into bf16 (k,k+1) pairs that
// match the B-fragment layout (lane n: col=n&15, kbase=(n<16?0:8)).
// ---------------------------------------------------------------------------
__global__ __launch_bounds__(256) void qkv_wmma_kernel(
    const float* __restrict__ x,
    const float* __restrict__ Wq, const float* __restrict__ bq,
    const float* __restrict__ Wk, const float* __restrict__ bk,
    const float* __restrict__ Wv, const float* __restrict__ bv,
    float* __restrict__ xq, float* __restrict__ xk, float* __restrict__ xv)
{
  __shared__ float    wraw[3 * 64 * 64];   // 48 KB fp32 weight staging
  __shared__ unsigned wpk [3 * 32 * 64];   // 24 KB packed bf16 pairs
  const int tid = threadIdx.x;

  // ---- async bulk copy: 3 x 16 KB weights -> LDS (b128 per lane) ----------
  {
    const unsigned lbase = (unsigned)(unsigned long long)(const void*)wraw;
    #pragma unroll
    for (int it = 0; it < 12; ++it) {
      int e = it * 256 + tid;              // float4 index, 0..3071
      int m = e >> 10;                     // 1024 float4 per matrix
      int r = e & 1023;
      const float* W = (m == 0) ? Wq : ((m == 1) ? Wk : Wv);
      unsigned long long ga = (unsigned long long)(const void*)(W + r * 4);
      unsigned la = lbase + (unsigned)e * 16u;
      asm volatile("global_load_async_to_lds_b128 %0, %1, off"
                   :: "v"(la), "v"(ga) : "memory");
    }
    asm volatile("s_wait_asynccnt 0x0" ::: "memory");
  }
  __syncthreads();

  // ---- pack fp32 -> bf16 (k,k+1) pair dwords in B-fragment order ----------
  #pragma unroll
  for (int it = 0; it < 24; ++it) {
    int e  = it * 256 + tid;               // 0 .. 6143
    int m  = e >> 11;
    int r  = e & 2047;
    int k2 = r >> 6;
    int c  = r & 63;
    const float* Wl = wraw + m * 4096 + (2 * k2) * 64 + c;
    wpk[e] = pack_bf16x2(Wl[0], Wl[64]);
  }
  __syncthreads();

  const int wave = tid >> 5;
  const int lane = tid & 31;
  const int r0   = (blockIdx.x * 8 + wave) * 16;
  const int arow = lane & 15;
  const int akb  = (lane < 16) ? 0 : 8;

  // A fragments: 16x32 bf16, lane L holds row M=L&15, K-chunks per ISA layout
  Frag a[2];
  #pragma unroll
  for (int ks = 0; ks < 2; ++ks) {
    #pragma unroll
    for (int v = 0; v < 8; ++v) {
      int kk = (v < 4) ? (akb + 2 * v) : (16 + akb + 2 * (v - 4));
      const float2 g = *reinterpret_cast<const float2*>(
          x + (size_t)(r0 + arow) * CH + ks * 32 + kk);
      a[ks].u[v] = pack_bf16x2(g.x, g.y);
    }
  }

  const int coln   = lane & 15;
  const int bkb2   = (lane < 16) ? 0 : 16;
  const int rowoff = (lane < 16) ? 0 : 8;

  #pragma unroll
  for (int m = 0; m < 3; ++m) {
    const float* bias = (m == 0) ? bq : ((m == 1) ? bk : bv);
    float*       dst  = (m == 0) ? xq : ((m == 1) ? xk : xv);
    #pragma unroll
    for (int nt = 0; nt < 4; ++nt) {
      f32x8 acc = {0.f, 0.f, 0.f, 0.f, 0.f, 0.f, 0.f, 0.f};
      #pragma unroll
      for (int ks = 0; ks < 2; ++ks) {
        Frag b;
        int k2b = ks * 16 + (bkb2 >> 2);
        #pragma unroll
        for (int v = 0; v < 8; ++v)
          b.u[v] = wpk[m * 2048 + (k2b + v + (v < 4 ? 0 : 4)) * 64 + nt * 16 + coln];
        acc = __builtin_amdgcn_wmma_f32_16x16x32_bf16(
            false, a[ks].v, false, b.v, (short)0, acc, false, false);
      }
      const float bb = bias[nt * 16 + coln];
      #pragma unroll
      for (int r = 0; r < 8; ++r)
        dst[(size_t)(r0 + r + rowoff) * CH + nt * 16 + coln] = acc[r] + bb;
    }
  }
}

// ---------------------------------------------------------------------------
// Kernel B: fused gather + position MLP + attention logits (WMMA for the
// 16x64 @ 64x8 w1 GEMM) + bn2/relu + w2 + softmax + share-plane reduction.
// One wave per query point; lane L owns neighbor j = L & 15. All small
// parameter tables staged once per block into LDS with BN scales pre-folded.
// ---------------------------------------------------------------------------
__global__ __launch_bounds__(256) void attn_kernel(
    const float* __restrict__ p,   const int* __restrict__ nidx,
    const float* __restrict__ p1W, const float* __restrict__ p1b,
    const float* __restrict__ pg,  const float* __restrict__ pb,
    const float* __restrict__ p2W, const float* __restrict__ p2b,
    const float* __restrict__ g1,  const float* __restrict__ b1,
    const float* __restrict__ w1W, const float* __restrict__ w1b,
    const float* __restrict__ g2,  const float* __restrict__ b2,
    const float* __restrict__ w2W, const float* __restrict__ w2b,
    const float* __restrict__ xq,  const float* __restrict__ xk,
    const float* __restrict__ xv,  float* __restrict__ out)
{
  __shared__ float    h1s[8][NSZ * CSZ];   // per-wave 16x8 h1 staging (4 KB)
  __shared__ float    s1s[64], b1s[64], p2Ws[3 * 64], p2bs[64];
  __shared__ float    w2Ws[64], w2bs[8], s2s[8], b2s[8], w1bs[8];
  __shared__ float    p1Ws[9], p1bs[3], pgs[3], pbs[3];
  __shared__ unsigned w1pk[32 * 8];        // packed bf16 B-frag dwords of w1_W

  const int tid  = threadIdx.x;
  const int wave = tid >> 5;
  const int lane = tid & 31;
  const int qi   = blockIdx.x * 8 + wave;
  const float inv = rsqrtf(1.f + 1e-5f);   // BN with running_var = 1

  // ---- stage parameter tables (once per block) ----------------------------
  if (tid < 64) {
    s1s[tid]  = g1[tid] * inv;             // bn1 scale folded
    b1s[tid]  = b1[tid];
    p2bs[tid] = p2b[tid];
    w2Ws[tid] = w2W[tid];
  }
  if (tid < 192) p2Ws[tid] = p2W[tid];
  if (tid < 8) {
    w2bs[tid] = w2b[tid];
    s2s[tid]  = g2[tid] * inv;             // bn2 scale folded
    b2s[tid]  = b2[tid];
    w1bs[tid] = w1b[tid];
  }
  if (tid < 9) p1Ws[tid] = p1W[tid];
  if (tid < 3) { p1bs[tid] = p1b[tid]; pgs[tid] = pg[tid] * inv; pbs[tid] = pb[tid]; }
  {                                        // w1_W -> packed (k,k+1) pairs
    int k2 = tid >> 3, o = tid & 7;
    w1pk[tid] = pack_bf16x2(w1W[(2 * k2) * CSZ + o], w1W[(2 * k2 + 1) * CSZ + o]);
  }
  __syncthreads();

  const int j   = lane & 15;
  const int nbr = nidx[qi * NSZ + j];
  __builtin_prefetch(xv + (size_t)nbr * CH, 0, 0);   // gv row used after softmax

  // position MLP first stage: q3 = relu(bn_p((p[nbr]-p[qi]) @ p1_W + p1_b))
  float q3[3];
  {
    const float rx = p[(size_t)nbr * 3 + 0] - p[(size_t)qi * 3 + 0];
    const float ry = p[(size_t)nbr * 3 + 1] - p[(size_t)qi * 3 + 1];
    const float rz = p[(size_t)nbr * 3 + 2] - p[(size_t)qi * 3 + 2];
    #pragma unroll
    for (int c = 0; c < 3; ++c) {
      float t = rx * p1Ws[c] + ry * p1Ws[3 + c] + rz * p1Ws[6 + c] + p1bs[c];
      t = t * pgs[c] + pbs[c];
      q3[c] = fmaxf(t, 0.f);
    }
  }

  // A fragment: relu(bn1(gk - xq + pr)) as bf16, built directly in WMMA layout
  const int akb = (lane < 16) ? 0 : 8;
  Frag a[2];
  #pragma unroll
  for (int ks = 0; ks < 2; ++ks) {
    #pragma unroll
    for (int v = 0; v < 8; ++v) {
      int kk = (v < 4) ? (akb + 2 * v) : (16 + akb + 2 * (v - 4));
      int ch = ks * 32 + kk;
      const float2 gk = *(const float2*)(xk + (size_t)nbr * CH + ch);
      const float2 qq = *(const float2*)(xq + (size_t)qi  * CH + ch);
      float pr0 = q3[0]*p2Ws[ch]   + q3[1]*p2Ws[64+ch]   + q3[2]*p2Ws[128+ch]   + p2bs[ch];
      float pr1 = q3[0]*p2Ws[ch+1] + q3[1]*p2Ws[64+ch+1] + q3[2]*p2Ws[128+ch+1] + p2bs[ch+1];
      float w0  = gk.x - qq.x + pr0;
      float w1v = gk.y - qq.y + pr1;
      w0  = fmaxf(w0  * s1s[ch]   + b1s[ch],   0.f);
      w1v = fmaxf(w1v * s1s[ch+1] + b1s[ch+1], 0.f);
      a[ks].u[v] = pack_bf16x2(w0, w1v);
    }
  }

  // B fragment from packed w1_W, columns 8..15 zero-padded (selects, no branch)
  const int  o16    = lane & 15;
  const int  oc     = o16 & 7;
  const bool ovalid = (o16 < 8);
  const int  bkb2   = (lane < 16) ? 0 : 16;
  f32x8 acc = {0.f, 0.f, 0.f, 0.f, 0.f, 0.f, 0.f, 0.f};
  #pragma unroll
  for (int ks = 0; ks < 2; ++ks) {
    Frag b;
    int k2b = ks * 16 + (bkb2 >> 2);
    #pragma unroll
    for (int v = 0; v < 8; ++v) {
      unsigned pk = w1pk[(k2b + v + (v < 4 ? 0 : 4)) * CSZ + oc];
      b.u[v] = ovalid ? pk : 0u;
    }
    acc = __builtin_amdgcn_wmma_f32_16x16x32_bf16(
        false, a[ks].v, false, b.v, (short)0, acc, false, false);
  }

  // h1 = relu(bn2(acc + w1_b)) -> LDS (C-frag: lane n = col n&15, rows r+8*(n>=16))
  if (ovalid) {
    const float s2 = s2s[oc], bb2 = b2s[oc], w1bb = w1bs[oc];
    const int rowoff = (lane < 16) ? 0 : 8;
    #pragma unroll
    for (int r = 0; r < 8; ++r) {
      float h = acc[r] + w1bb;
      h1s[wave][(r + rowoff) * CSZ + oc] = fmaxf(h * s2 + bb2, 0.f);
    }
  }
  __syncthreads();

  // w2 GEMM (8x8) + softmax over the 16 neighbors (lane = neighbor, xor-reduce)
  float hv[8];
  #pragma unroll
  for (int cs = 0; cs < 8; ++cs) hv[cs] = h1s[wave][j * CSZ + cs];
  float wgt[8];
  #pragma unroll
  for (int o = 0; o < 8; ++o) {
    float h2 = w2bs[o];
    #pragma unroll
    for (int cs = 0; cs < 8; ++cs) h2 += hv[cs] * w2Ws[cs * CSZ + o];
    float mx = h2;
    #pragma unroll
    for (int m = 1; m <= 8; m <<= 1) mx = fmaxf(mx, __shfl_xor(mx, m));
    float e = __expf(h2 - mx);
    float sm = e;
    #pragma unroll
    for (int m = 1; m <= 8; m <<= 1) sm += __shfl_xor(sm, m);
    wgt[o] = e / sm;
  }

  // out[qi, s*8+o] = sum_j (gv + pr)[s*8+o] * wgt[j][o]
  #pragma unroll
  for (int s = 0; s < 8; ++s) {
    const float* gv = xv + (size_t)nbr * CH + s * 8;
    float t[8];
    #pragma unroll
    for (int o = 0; o < 8; ++o) {
      int ch = s * 8 + o;
      float pr = q3[0]*p2Ws[ch] + q3[1]*p2Ws[64+ch] + q3[2]*p2Ws[128+ch] + p2bs[ch];
      t[o] = (gv[o] + pr) * wgt[o];
    }
    #pragma unroll
    for (int o = 0; o < 8; ++o)
      #pragma unroll
      for (int m = 1; m <= 8; m <<= 1) t[o] += __shfl_xor(t[o], m);
    if (lane == 0) {
      *reinterpret_cast<float4*>(out + (size_t)qi * CH + s * 8) =
          make_float4(t[0], t[1], t[2], t[3]);
      *reinterpret_cast<float4*>(out + (size_t)qi * CH + s * 8 + 4) =
          make_float4(t[4], t[5], t[6], t[7]);
    }
  }
}

// ---------------------------------------------------------------------------
extern "C" void kernel_launch(void* const* d_in, const int* in_sizes, int n_in,
                              void* d_out, int out_size, void* d_ws, size_t ws_size,
                              hipStream_t stream) {
  const float* p    = (const float*)d_in[0];
  const float* x    = (const float*)d_in[1];
  const int*   nidx = (const int*)  d_in[2];
  const float* Wq   = (const float*)d_in[3];
  const float* bq   = (const float*)d_in[4];
  const float* Wk   = (const float*)d_in[5];
  const float* bk   = (const float*)d_in[6];
  const float* Wv   = (const float*)d_in[7];
  const float* bv   = (const float*)d_in[8];
  const float* p1W  = (const float*)d_in[9];
  const float* p1b  = (const float*)d_in[10];
  const float* pg   = (const float*)d_in[11];
  const float* pbb  = (const float*)d_in[12];
  const float* p2W  = (const float*)d_in[13];
  const float* p2b  = (const float*)d_in[14];
  const float* g1   = (const float*)d_in[15];
  const float* b1   = (const float*)d_in[16];
  const float* w1W  = (const float*)d_in[17];
  const float* w1b  = (const float*)d_in[18];
  const float* g2   = (const float*)d_in[19];
  const float* b2   = (const float*)d_in[20];
  const float* w2W  = (const float*)d_in[21];
  const float* w2b  = (const float*)d_in[22];

  float* xq = (float*)d_ws;                  // 3 * N*C floats = 48 MB scratch
  float* xk = xq + (size_t)NPTS * CH;
  float* xv = xk + (size_t)NPTS * CH;

  qkv_wmma_kernel<<<NPTS / 128, 256, 0, stream>>>(x, Wq, bq, Wk, bk, Wv, bv,
                                                  xq, xk, xv);
  attn_kernel<<<NPTS / 8, 256, 0, stream>>>(p, nidx, p1W, p1b, pg, pbb, p2W, p2b,
                                            g1, b1, w1W, w1b, g2, b2, w2W, w2b,
                                            xq, xk, xv, (float*)d_out);
}
